// Head_63599875719322
// MI455X (gfx1250) — compile-verified, matrix operand order checked by reading, not run on hardware
//
#include <hip/hip_runtime.h>
#include <hip/hip_bf16.h>

// ---------------------------------------------------------------------------
// Fused attention head for MI455X (gfx1250, wave32, WMMA).
//
// Roofline: x read = 96MB (~4.2us @23.3TB/s); total math ~18.3 GFLOP done on
// v_wmma_f32_16x16x32_f16. q/k/vT f16 intermediates (12MB) stay in the 192MB
// L2; the 32x1024x1024 attention matrix never exists in memory (flash-style
// online softmax per 16-query wave tile).
// ---------------------------------------------------------------------------

typedef __attribute__((ext_vector_type(16))) _Float16 v16h;
typedef __attribute__((ext_vector_type(8)))  _Float16 v8h;
typedef __attribute__((ext_vector_type(8)))  float    v8f;
typedef __attribute__((ext_vector_type(4)))  float    v4f;

#define N_EMBD 768
#define HEADD  64
#define BATCH  32
#define SEQ    1024
#define MROWS  (BATCH * SEQ)            // 32768 token rows
#define SCALE_Q 0.03608439182435161f    // 768^-0.5
#define LN10K_64 0.14391156856871347f   // ln(10000)/64

// ---- per-wave LDS fence: DS ops are in-order per wave; keep compiler honest
__device__ __forceinline__ void lds_fence() {
    __builtin_amdgcn_wave_barrier();
    asm volatile("s_wait_dscnt 0" ::: "memory");
    __builtin_amdgcn_wave_barrier();
}

__device__ __forceinline__ v8f wmma16(v16h a, v16h b, v8f c) {
    // D = A(16x32 f16) * B(32x16 f16) + C(16x16 f32)
    return __builtin_amdgcn_wmma_f32_16x16x32_f16(
        /*neg_a=*/false, a, /*neg_b=*/false, b,
        /*c_mod=*/(short)0, c, /*reuse_a=*/false, /*reuse_b=*/false);
}

// 16-bit A/B fragment (ISA 7.12.2): lanes 0-15 hold row (n0+lane), K {0..7,16..23};
// lanes 16-31 hold row (n0+lane-16), K {8..15,24..31}. Source is row-major f16
// with leading dim ld, K offset k0. Two contiguous 16B loads per lane.
__device__ __forceinline__ v16h load_frag_f16(const _Float16* p, int n0, int k0, int ld) {
    const int lane = threadIdx.x & 31;
    const int n    = lane & 15;
    const int ks   = (lane < 16) ? 0 : 8;
    const _Float16* base = p + (size_t)(n0 + n) * ld + k0 + ks;
    v8h lo = *(const v8h*)(base);
    v8h hi = *(const v8h*)(base + 16);
    v16h r;
#pragma unroll
    for (int i = 0; i < 8; ++i) { r[i] = lo[i]; r[8 + i] = hi[i]; }
    return r;
}

// Same fragment but sourced from fp32 (for the x activations), converted on load.
__device__ __forceinline__ v16h load_frag_f32(const float* p, int n0, int k0, int ld) {
    const int lane = threadIdx.x & 31;
    const int n    = lane & 15;
    const int ks   = (lane < 16) ? 0 : 8;
    const float* base = p + (size_t)(n0 + n) * ld + k0 + ks;
    v4f a0 = *(const v4f*)(base);
    v4f a1 = *(const v4f*)(base + 4);
    v4f b0 = *(const v4f*)(base + 16);
    v4f b1 = *(const v4f*)(base + 20);
    v16h r;
#pragma unroll
    for (int i = 0; i < 4; ++i) {
        r[i]      = (_Float16)a0[i];
        r[4 + i]  = (_Float16)a1[i];
        r[8 + i]  = (_Float16)b0[i];
        r[12 + i] = (_Float16)b1[i];
    }
    return r;
}

// ---------------------------------------------------------------------------
// Kernel 0: W[768][64] fp32 -> Wt[64][768] f16 (x3), so B-fragments are
// two contiguous ds/global b128 loads per lane.
// ---------------------------------------------------------------------------
__global__ __launch_bounds__(256) void wt_prep_kernel(
    const float* __restrict__ Wq, const float* __restrict__ Wk,
    const float* __restrict__ Wv, _Float16* __restrict__ Wt) {
    int idx = blockIdx.x * 256 + threadIdx.x;       // exactly 3*768*64 threads
    int w   = idx / (N_EMBD * HEADD);
    int rme = idx % (N_EMBD * HEADD);
    int kk  = rme >> 6;                             // k index 0..767
    int nn  = rme & 63;                             // n index 0..63
    const float* W = (w == 0) ? Wq : ((w == 1) ? Wk : Wv);
    Wt[(size_t)w * (HEADD * N_EMBD) + (size_t)nn * N_EMBD + kk] = (_Float16)W[rme];
}

// ---------------------------------------------------------------------------
// Kernel 1: q/k/v projection + RoPE. One wave per 16 token rows.
// 24 k-steps x (1 A-frag + 12 B-frags + 12 WMMAs). q gets the softmax scale
// folded in. q,k stored row-major f16 via LDS repack (coalesced b128 stores);
// v stored transposed vT[b][d][t] with packed 16B stores.
// ---------------------------------------------------------------------------
__global__ __launch_bounds__(256) void qkv_rope_kernel(
    const float* __restrict__ x, const _Float16* __restrict__ Wt,
    _Float16* __restrict__ q, _Float16* __restrict__ k,
    _Float16* __restrict__ vT) {
    __shared__ _Float16 lds[8 * 16 * HEADD];        // 2KB per wave
    const int wave  = (blockIdx.x * blockDim.x + threadIdx.x) >> 5;
    const int wlocal = threadIdx.x >> 5;
    const int lane  = threadIdx.x & 31;
    const int n     = lane & 15;
    const int half  = lane >> 4;
    const int m0    = wave * 16;                    // global token-row base
    const int b     = m0 >> 10;
    const int t0    = m0 & 1023;
    _Float16* lp = lds + wlocal * 16 * HEADD;

    const _Float16* Wtq = Wt;
    const _Float16* Wtk = Wt + 1 * HEADD * N_EMBD;
    const _Float16* Wtv = Wt + 2 * HEADD * N_EMBD;

    v8f accq[4], acck[4], accv[4];
    const v8f zf = {0.f, 0.f, 0.f, 0.f, 0.f, 0.f, 0.f, 0.f};
#pragma unroll
    for (int nt = 0; nt < 4; ++nt) { accq[nt] = zf; acck[nt] = zf; accv[nt] = zf; }

    for (int kk = 0; kk < N_EMBD; kk += 32) {
        v16h ax = load_frag_f32(x, m0, kk, N_EMBD);
#pragma unroll
        for (int nt = 0; nt < 4; ++nt) {
            accq[nt] = wmma16(ax, load_frag_f16(Wtq, nt * 16, kk, N_EMBD), accq[nt]);
            acck[nt] = wmma16(ax, load_frag_f16(Wtk, nt * 16, kk, N_EMBD), acck[nt]);
            accv[nt] = wmma16(ax, load_frag_f16(Wtv, nt * 16, kk, N_EMBD), accv[nt]);
        }
    }

    // ---- RoPE: pair (2i,2i+1) lives in adjacent lanes (C-layout: lane = column)
#pragma unroll
    for (int nt = 0; nt < 4; ++nt) {
        const int dcol = nt * 16 + n;
        const float freq = __expf(-(float)(dcol & ~1) * LN10K_64);
        const float sgn  = (dcol & 1) ? 1.f : -1.f;
#pragma unroll
        for (int r = 0; r < 8; ++r) {
            float pos = (float)(t0 + r + 8 * half);
            float s, c;
            __sincosf(pos * freq, &s, &c);
            float vq = accq[nt][r], pq = __shfl_xor(vq, 1, 32);
            float vk = acck[nt][r], pk = __shfl_xor(vk, 1, 32);
            accq[nt][r] = (vq * c + sgn * pq * s) * SCALE_Q;  // fold softmax scale
            acck[nt][r] = (vk * c + sgn * pk * s);
        }
    }

    // ---- q then k: scatter C-layout into LDS, read back rows, coalesced stores
#pragma unroll
    for (int which = 0; which < 2; ++which) {
        v8f* acc = which ? acck : accq;
        _Float16* gout = which ? k : q;
#pragma unroll
        for (int nt = 0; nt < 4; ++nt)
#pragma unroll
            for (int r = 0; r < 8; ++r)
                lp[(r + 8 * half) * HEADD + nt * 16 + n] = (_Float16)acc[nt][r];
        lds_fence();
        {
            const int row = lane >> 1;
            const int co  = (lane & 1) * 32;
#pragma unroll
            for (int j = 0; j < 4; ++j) {
                v8h d = *(const v8h*)(lp + row * HEADD + co + j * 8);
                *(v8h*)(gout + (size_t)(m0 + row) * HEADD + co + j * 8) = d;
            }
        }
        lds_fence();
    }

    // ---- v: store transposed vT[b][d][t]; 8 consecutive t per lane -> 16B store
#pragma unroll
    for (int nt = 0; nt < 4; ++nt) {
        v8h pk;
#pragma unroll
        for (int r = 0; r < 8; ++r) pk[r] = (_Float16)accv[nt][r];
        size_t d = (size_t)b * HEADD + nt * 16 + n;
        *(v8h*)(vT + d * SEQ + t0 + 8 * half) = pk;
    }
}

// ---------------------------------------------------------------------------
// Kernel 2: flash attention. One wave per 16-query tile; loop over 32-key
// chunks (wave-uniform causal bound -> EXEC all-1s at every WMMA).
// S = Q K^T (4 wmma/chunk), online softmax via 16-lane shfl_xor butterflies,
// P repacked C->A layout through 1KB per-wave LDS, P V via vT (4 wmma/chunk).
// ---------------------------------------------------------------------------
__global__ __launch_bounds__(256) void flash_attn_kernel(
    const _Float16* __restrict__ q, const _Float16* __restrict__ k,
    const _Float16* __restrict__ vT, float* __restrict__ out) {
    __shared__ _Float16 ldsP[8 * 16 * 32];          // 1KB per wave
    const int wave  = (blockIdx.x * blockDim.x + threadIdx.x) >> 5;
    const int lane  = threadIdx.x & 31;
    const int n     = lane & 15;
    const int half  = lane >> 4;
    const int qrow0 = wave * 16;
    const int b     = qrow0 >> 10;
    const int t0    = qrow0 & 1023;
    _Float16* lp = ldsP + (threadIdx.x >> 5) * 512;
    const _Float16* vb = vT + (size_t)b * HEADD * SEQ;

    v16h aq0 = load_frag_f16(q, qrow0, 0, HEADD);
    v16h aq1 = load_frag_f16(q, qrow0, 32, HEADD);

    v8f acc[4];
    const v8f zf = {0.f, 0.f, 0.f, 0.f, 0.f, 0.f, 0.f, 0.f};
#pragma unroll
    for (int nt = 0; nt < 4; ++nt) acc[nt] = zf;
    float mrow[8], lrow[8];
#pragma unroll
    for (int r = 0; r < 8; ++r) { mrow[r] = -3.0e38f; lrow[r] = 0.f; }

    const int nch = (t0 >> 5) + 1;
    for (int c2 = 0; c2 < nch; ++c2) {
        const int cb = c2 << 5;
        // ---- S = Q K^T for 32 keys (two 16x16 tiles)
        v8f S0 = zf, S1 = zf;
        S0 = wmma16(aq0, load_frag_f16(k, (b << 10) + cb, 0, HEADD), S0);
        S0 = wmma16(aq1, load_frag_f16(k, (b << 10) + cb, 32, HEADD), S0);
        S1 = wmma16(aq0, load_frag_f16(k, (b << 10) + cb + 16, 0, HEADD), S1);
        S1 = wmma16(aq1, load_frag_f16(k, (b << 10) + cb + 16, 32, HEADD), S1);

        if (c2 == nch - 1) {                        // causal mask (diag chunk only)
            const int key0 = cb + n, key1 = key0 + 16;
#pragma unroll
            for (int r = 0; r < 8; ++r) {
                const int row = t0 + r + 8 * half;
                if (key0 > row) S0[r] = -1.0e30f;
                if (key1 > row) S1[r] = -1.0e30f;
            }
        }

        // ---- online softmax; row's 16 values live across one 16-lane group
#pragma unroll
        for (int r = 0; r < 8; ++r) {
            float mx = fmaxf(S0[r], S1[r]);
#pragma unroll
            for (int off = 1; off < 16; off <<= 1) mx = fmaxf(mx, __shfl_xor(mx, off, 32));
            const float mn = fmaxf(mrow[r], mx);
            const float sc = __expf(mrow[r] - mn);
            const float p0 = __expf(S0[r] - mn);
            const float p1 = __expf(S1[r] - mn);
            float rs = p0 + p1;
#pragma unroll
            for (int off = 1; off < 16; off <<= 1) rs += __shfl_xor(rs, off, 32);
            lrow[r] = lrow[r] * sc + rs;
            mrow[r] = mn;
#pragma unroll
            for (int nt = 0; nt < 4; ++nt) acc[nt][r] *= sc;
            lp[(r + 8 * half) * 32 + n]      = (_Float16)p0;
            lp[(r + 8 * half) * 32 + n + 16] = (_Float16)p1;
        }
        lds_fence();
        v16h pa = load_frag_f16(lp, 0, 0, 32);      // A-layout reload (ds_load_b128)
        // ---- O += P V  (V^T gives contiguous B-fragments)
#pragma unroll
        for (int nt = 0; nt < 4; ++nt)
            acc[nt] = wmma16(pa, load_frag_f16(vb, nt * 16, cb, SEQ), acc[nt]);
        lds_fence();                                 // before next chunk reuses lp
    }

    // ---- epilogue: normalize and store fp32 output (coalesced b32 across lanes)
#pragma unroll
    for (int r = 0; r < 8; ++r) {
        const float inv = 1.0f / lrow[r];
        const size_t row = (size_t)(qrow0 + r + 8 * half);
#pragma unroll
        for (int nt = 0; nt < 4; ++nt)
            out[row * HEADD + nt * 16 + n] = acc[nt][r] * inv;
    }
}

// ---------------------------------------------------------------------------
extern "C" void kernel_launch(void* const* d_in, const int* in_sizes, int n_in,
                              void* d_out, int out_size, void* d_ws, size_t ws_size,
                              hipStream_t stream) {
    const float* x  = (const float*)d_in[0];
    const float* Wq = (const float*)d_in[1];
    const float* Wk = (const float*)d_in[2];
    const float* Wv = (const float*)d_in[3];
    float* out = (float*)d_out;

    // workspace layout (f16): Wt[3][64][768] | q[32768][64] | k[32768][64] | vT[32][64][1024]
    unsigned char* ws = (unsigned char*)d_ws;
    const size_t WT_BYTES = (size_t)3 * HEADD * N_EMBD * 2;   //   294912
    const size_t QK_BYTES = (size_t)MROWS * HEADD * 2;        //  4194304
    _Float16* Wt = (_Float16*)(ws);
    _Float16* q  = (_Float16*)(ws + WT_BYTES);
    _Float16* kk = (_Float16*)(ws + WT_BYTES + QK_BYTES);
    _Float16* vT = (_Float16*)(ws + WT_BYTES + 2 * QK_BYTES);

    // 0) weight transpose/convert: 147456 elems, exact grid
    wt_prep_kernel<<<(3 * N_EMBD * HEADD) / 256, 256, 0, stream>>>(Wq, Wk, Wv, Wt);

    // 1) QKV + RoPE: 2048 waves, 8 waves/block
    qkv_rope_kernel<<<MROWS / 128, 256, 0, stream>>>(x, Wt, q, kk, vT);

    // 2) flash attention: 2048 query tiles, 8 waves/block
    flash_attn_kernel<<<(MROWS / 16) / 8, 256, 0, stream>>>(q, kk, vT, out);
}
